// ReDrafterHead_56315611185403
// MI455X (gfx1250) — matrix operand order repaired and checked by reading, not compiled, not run on hardware
//
#include <hip/hip_runtime.h>
#include <hip/hip_bf16.h>

// ---------------------------------------------------------------------------
// ReDrafter head for MI455X (gfx1250, wave32, WMMA + TDM).
//   1) h0 = hidden @ ip_w.T + ip_b           (bf16-split WMMA GEMM)
//   2) gather all 4 step inputs from embed   (independent of recurrence)
//   3) 8 sequential fused GRU-cell kernels   (bf16-split WMMA + exp epilogue);
//      layer-1 epilogue emits h2 as bf16 hi/lo planes for the logits GEMM
//   4) ONE batched logits GEMM (256 x 50257 x 512):
//      per-WG 32-col out_w strip (64KB fp32) DMA'd into LDS by the Tensor Data
//      Mover, converted once to bf16 hi/lo in LDS, then all 8 waves run a pure
//      ds_load -> v_wmma pipeline (no conversion VALU in the hot loop).
//      128KB LDS/WG -> 2 WGs per WGP so TDM of one WG overlaps compute of the
//      other. out_w is read from HBM exactly once (103MB ~ 4.4us floor).
// Precision: fp32 emulated via bf16 hi/lo split: a*b ~= ah*bh + ah*bl + al*bh
// ---------------------------------------------------------------------------

typedef __attribute__((ext_vector_type(16))) __bf16 v16bf;
typedef __attribute__((ext_vector_type(8)))  __bf16 v8bf;
typedef __attribute__((ext_vector_type(4)))  __bf16 v4bf;
typedef __attribute__((ext_vector_type(8)))  float  v8f;
typedef unsigned int u32x4 __attribute__((ext_vector_type(4)));
typedef int          i32x4 __attribute__((ext_vector_type(4)));
typedef int          i32x8 __attribute__((ext_vector_type(8)));

#define DH     512
#define HIDDEN 4096
#define VOCAB  50257
#define BATCH  64
#define NDRAFT 4

static __device__ __forceinline__ void st4(float* v, float4 a) {
  v[0] = a.x; v[1] = a.y; v[2] = a.z; v[3] = a.w;
}

// fp32 -> bf16 hi/lo split of a 16-element fragment
static __device__ __forceinline__ void split_bf16(const float* vals, v16bf& hi, v16bf& lo) {
#pragma unroll
  for (int j = 0; j < 16; ++j) {
    float f  = vals[j];
    __bf16 h = (__bf16)f;
    hi[j] = h;
    lo[j] = (__bf16)(f - (float)h);
  }
}

// A fragment from fp32: 16x32 (MxK). lane<16 -> row M=lane, K {k0..k0+15}
//                                    lane>=16 -> row M=lane-16, K {k0+16..k0+31}
static __device__ __forceinline__ void load_a_frag(const float* __restrict__ A, int lda,
                                                   int m0, int k0, int lane,
                                                   v16bf& hi, v16bf& lo) {
  int row = m0 + (lane & 15);
  int kb  = k0 + ((lane >> 4) << 4);
  const float* p = A + (size_t)row * lda + kb;
  float vals[16];
  st4(vals + 0,  *(const float4*)(p + 0));
  st4(vals + 4,  *(const float4*)(p + 4));
  st4(vals + 8,  *(const float4*)(p + 8));
  st4(vals + 12, *(const float4*)(p + 12));
  split_bf16(vals, hi, lo);
}

// A fragment straight from a pre-split bf16 plane (no conversion VALU).
static __device__ __forceinline__ v16bf load_a_frag_bf16(const __bf16* __restrict__ A,
                                                         int m0, int k0, int lane) {
  int row = m0 + (lane & 15);
  int kb  = k0 + ((lane >> 4) << 4);
  const __bf16* p = A + (size_t)row * DH + kb;
  v8bf a0 = *(const v8bf*)(p);
  v8bf a1 = *(const v8bf*)(p + 8);
  return __builtin_shufflevector(a0, a1, 0, 1, 2, 3, 4, 5, 6, 7,
                                          8, 9, 10, 11, 12, 13, 14, 15);
}

// B fragment from fp32 (N x K row-major, NT gemm).
// lane<16 -> col N=lane, K {k0..k0+15}; lane>=16 -> col N=lane-16, K {k0+16..k0+31}
static __device__ __forceinline__ void load_b_frag(const float* __restrict__ B, int ldb,
                                                   int n0, int k0, int lane, int nmax,
                                                   v16bf& hi, v16bf& lo) {
  int col = n0 + (lane & 15);
  if (col > nmax) col = nmax;
  int kb  = k0 + ((lane >> 4) << 4);
  const float* p = B + (size_t)col * ldb + kb;
  float vals[16];
  st4(vals + 0,  *(const float4*)(p + 0));
  st4(vals + 4,  *(const float4*)(p + 4));
  st4(vals + 8,  *(const float4*)(p + 8));
  st4(vals + 12, *(const float4*)(p + 12));
  split_bf16(vals, hi, lo);
}

// 3-term bf16 emulation of an fp32 MAC tile: acc += A*B
static __device__ __forceinline__ v8f wmma3(v8f acc,
                                            const v16bf& ah, const v16bf& al,
                                            const v16bf& bh, const v16bf& bl) {
  acc = __builtin_amdgcn_wmma_f32_16x16x32_bf16(false, ah, false, bh, (short)0, acc, false, false);
  acc = __builtin_amdgcn_wmma_f32_16x16x32_bf16(false, ah, false, bl, (short)0, acc, false, false);
  acc = __builtin_amdgcn_wmma_f32_16x16x32_bf16(false, al, false, bh, (short)0, acc, false, false);
  return acc;
}

static __device__ __forceinline__ float sigmoidf_(float x) {
  return 1.0f / (1.0f + __expf(-x));
}
static __device__ __forceinline__ float tanhf_(float x) {
  float e = __expf(-2.0f * x);
  return (1.0f - e) / (1.0f + e);
}

// ---------------------------------------------------------------------------
// Kernel 1: h0 = hidden(64x4096) @ ip_w(512x4096)^T + ip_b  -> s0 and s1
// ---------------------------------------------------------------------------
__global__ __launch_bounds__(256) void k_h0(const float* __restrict__ X,
                                            const float* __restrict__ W,
                                            const float* __restrict__ bias,
                                            float* __restrict__ s0,
                                            float* __restrict__ s1) {
  const int lane = threadIdx.x & 31;
  const int wave = threadIdx.x >> 5;
  const int m0 = (wave & 1) * 32;
  const int n0 = blockIdx.x * 128 + (wave >> 1) * 32;

  v8f acc[2][2];
#pragma unroll
  for (int a = 0; a < 2; ++a)
#pragma unroll
    for (int b = 0; b < 2; ++b)
#pragma unroll
      for (int i = 0; i < 8; ++i) acc[a][b][i] = 0.0f;

  for (int k0 = 0; k0 < HIDDEN; k0 += 32) {
    v16bf ah[2], al[2], bh[2], bl[2];
#pragma unroll
    for (int mt = 0; mt < 2; ++mt) load_a_frag(X, HIDDEN, m0 + 16 * mt, k0, lane, ah[mt], al[mt]);
#pragma unroll
    for (int nt = 0; nt < 2; ++nt) load_b_frag(W, HIDDEN, n0 + 16 * nt, k0, lane, DH - 1, bh[nt], bl[nt]);
#pragma unroll
    for (int mt = 0; mt < 2; ++mt)
#pragma unroll
      for (int nt = 0; nt < 2; ++nt) acc[mt][nt] = wmma3(acc[mt][nt], ah[mt], al[mt], bh[nt], bl[nt]);
  }

#pragma unroll
  for (int mt = 0; mt < 2; ++mt)
#pragma unroll
    for (int nt = 0; nt < 2; ++nt) {
      int col = n0 + nt * 16 + (lane & 15);
      float bv = bias[col];
#pragma unroll
      for (int i = 0; i < 8; ++i) {
        int row = m0 + mt * 16 + i + ((lane >> 4) << 3);
        float v = acc[mt][nt][i] + bv;
        s0[row * DH + col] = v;
        s1[row * DH + col] = v;
      }
    }
}

// ---------------------------------------------------------------------------
// Kernel 2: xAll[s][b][d]: step 0 = zeros; step s = embed[target_ids[b, s-1]]
// ---------------------------------------------------------------------------
__global__ void k_gather(const float* __restrict__ embed,
                         const int* __restrict__ tids,
                         float* __restrict__ xAll) {
  int idx = blockIdx.x * blockDim.x + threadIdx.x;
  if (idx >= NDRAFT * BATCH * DH) return;
  int d = idx & (DH - 1);
  int b = (idx >> 9) & (BATCH - 1);
  int s = idx >> 15;
  float v = 0.0f;
  if (s > 0) {
    int t = tids[b * NDRAFT + (s - 1)];
    v = embed[(size_t)t * DH + d];
  }
  xAll[idx] = v;
}

// ---------------------------------------------------------------------------
// Kernel 3: fused GRU cell (6 gate GEMM tiles per wave + sigmoid/tanh epilogue)
// Layer-1 launches also bank h2 as bf16 hi/lo planes for the logits GEMM.
// ---------------------------------------------------------------------------
__global__ __launch_bounds__(256) void k_gru(const float* __restrict__ X,
                                             const float* __restrict__ H,
                                             const float* __restrict__ Wih,
                                             const float* __restrict__ Whh,
                                             const float* __restrict__ bih,
                                             const float* __restrict__ bhh,
                                             float* __restrict__ Hout,
                                             __bf16* __restrict__ H2hi,
                                             __bf16* __restrict__ H2lo) {
  const int lane = threadIdx.x & 31;
  const int wave = threadIdx.x >> 5;
  const int m0 = (wave & 3) * 16;
  const int n0 = blockIdx.x * 32 + (wave >> 2) * 16;

  v8f gi[3], gh[3];
#pragma unroll
  for (int g = 0; g < 3; ++g)
#pragma unroll
    for (int i = 0; i < 8; ++i) { gi[g][i] = 0.0f; gh[g][i] = 0.0f; }

  for (int k0 = 0; k0 < DH; k0 += 32) {
    v16bf xh, xl, hh, hl;
    load_a_frag(X, DH, m0, k0, lane, xh, xl);
    load_a_frag(H, DH, m0, k0, lane, hh, hl);
#pragma unroll
    for (int g = 0; g < 3; ++g) {
      v16bf wih_h, wih_l, whh_h, whh_l;
      load_b_frag(Wih, DH, g * DH + n0, k0, lane, 3 * DH - 1, wih_h, wih_l);
      load_b_frag(Whh, DH, g * DH + n0, k0, lane, 3 * DH - 1, whh_h, whh_l);
      gi[g] = wmma3(gi[g], xh, xl, wih_h, wih_l);
      gh[g] = wmma3(gh[g], hh, hl, whh_h, whh_l);
    }
  }

  const int col = n0 + (lane & 15);
  const float b_ir = bih[col],          b_hr = bhh[col];
  const float b_iz = bih[DH + col],     b_hz = bhh[DH + col];
  const float b_in = bih[2 * DH + col], b_hn = bhh[2 * DH + col];

#pragma unroll
  for (int i = 0; i < 8; ++i) {
    int row = m0 + i + ((lane >> 4) << 3);
    float r = sigmoidf_(gi[0][i] + gh[0][i] + b_ir + b_hr);
    float z = sigmoidf_(gi[1][i] + gh[1][i] + b_iz + b_hz);
    float n = tanhf_((gi[2][i] + b_in) + r * (gh[2][i] + b_hn));
    float ho = H[row * DH + col];
    float v = (1.0f - z) * n + z * ho;
    Hout[row * DH + col] = v;
    if (H2hi) {
      __bf16 h = (__bf16)v;
      H2hi[row * DH + col] = h;
      H2lo[row * DH + col] = (__bf16)(v - (float)h);
    }
  }
}

// ---------------------------------------------------------------------------
// Kernel 4: batched logits GEMM D(256 x 50257) = H2all(256x512) @ out_w^T.
// Per-WG: TDM DMAs a 32-col out_w strip (32x512 fp32 = 64KB) into LDS (OOB
// rows zero-filled via tensor_dim1), one cooperative pass converts it to
// bf16 hi/lo LDS planes, then 8 waves (each 32 M-rows x 32 N-cols) run a
// conversion-free ds_load -> v_wmma K-loop. 128KB dynamic LDS.
// ---------------------------------------------------------------------------
__global__ __launch_bounds__(256) void k_logits(const __bf16* __restrict__ Ahi,
                                                const __bf16* __restrict__ Alo,
                                                const float* __restrict__ B,
                                                float* __restrict__ Out) {
  extern __shared__ char smem[];
  float*  rawB = (float*)smem;                        // 32*512 fp32  (64KB)
  __bf16* Bhi  = (__bf16*)(smem + 32 * DH * 4);       // 32*512 bf16  (32KB)
  __bf16* Blo  = (__bf16*)(smem + 32 * DH * 6);       // 32*512 bf16  (32KB)

  const int lane = threadIdx.x & 31;
  const int wave = threadIdx.x >> 5;
  const int m0 = wave * 32;
  const int n0 = blockIdx.x * 32;

#if __has_builtin(__builtin_amdgcn_tensor_load_to_lds) && __has_builtin(__builtin_amdgcn_s_wait_tensorcnt)
  if (wave == 0) {
    // Tensor DMA descriptor (D#), ISA ch.8: 2D tile 512(K) x 32(rows), fp32.
    unsigned long long ga = (unsigned long long)(const void*)(B + (size_t)n0 * DH);
    unsigned int rows_left = (unsigned int)(VOCAB - n0);   // OOB rows -> zero fill
    if (rows_left > 65535u) rows_left = 65535u;
    u32x4 g0;
    g0[0] = 1u;                                            // count=1 (valid), user mode
    g0[1] = 0u;                                            // lds_addr: dynamic LDS base
    g0[2] = (unsigned int)ga;                              // global_addr[31:0]
    g0[3] = ((unsigned int)(ga >> 32) & 0x1FFFFFFu) | 0x80000000u; // addr[56:32] | type=2
    i32x8 g1;
    g1[0] = (int)(2u << 16);                               // data_size = 4B
    g1[1] = (int)(512u << 16);                             // tensor_dim0[15:0] = 512 (K)
    g1[2] = (int)(rows_left << 16);                        // tensor_dim1[15:0]
    g1[3] = (int)(512u << 16);                             // tile_dim0 = 512
    g1[4] = 32;                                            // tile_dim1 = 32 rows
    g1[5] = 512;                                           // tensor_dim0_stride = 512
    g1[6] = 0;
    g1[7] = 0;
    i32x4 z4; z4[0] = 0; z4[1] = 0; z4[2] = 0; z4[3] = 0;
#if defined(__clang_major__) && (__clang_major__ >= 23)
    i32x8 z8;
#pragma unroll
    for (int i = 0; i < 8; ++i) z8[i] = 0;
    __builtin_amdgcn_tensor_load_to_lds(g0, g1, z4, z4, z8, 0);
#else
    __builtin_amdgcn_tensor_load_to_lds(g0, g1, z4, z4, 0);
#endif
    __builtin_amdgcn_s_wait_tensorcnt(0);
  }
  __syncthreads();
#else
  // Fallback: cooperative staged load of the strip (zero-fill OOB rows).
  for (int i = threadIdx.x; i < 32 * DH / 4; i += 256) {
    int idx = i * 4;
    int r = idx >> 9;
    int c = idx & (DH - 1);
    int col = n0 + r;
    float4 v = make_float4(0.f, 0.f, 0.f, 0.f);
    if (col < VOCAB) v = *(const float4*)(B + (size_t)col * DH + c);
    *(float4*)(rawB + idx) = v;
  }
  __syncthreads();
#endif

  // One cooperative fp32 -> bf16 hi/lo conversion pass (amortized over 8 waves
  // x 16 K-iters of reuse; hot loop below has zero conversion VALU).
  for (int i = threadIdx.x; i < 32 * DH / 4; i += 256) {
    int idx = i * 4;
    float4 f = *(const float4*)(rawB + idx);
    float vals[4] = { f.x, f.y, f.z, f.w };
    v4bf h4, l4;
#pragma unroll
    for (int j = 0; j < 4; ++j) {
      __bf16 h = (__bf16)vals[j];
      h4[j] = h;
      l4[j] = (__bf16)(vals[j] - (float)h);
    }
    *(v4bf*)(Bhi + idx) = h4;
    *(v4bf*)(Blo + idx) = l4;
  }
  __syncthreads();

  v8f acc[2][2];
#pragma unroll
  for (int a = 0; a < 2; ++a)
#pragma unroll
    for (int b = 0; b < 2; ++b)
#pragma unroll
      for (int i = 0; i < 8; ++i) acc[a][b][i] = 0.0f;

  for (int k0 = 0; k0 < DH; k0 += 32) {
    const int kb = k0 + ((lane >> 4) << 4);
    v16bf ah[2], al[2], bh[2], bl[2];
#pragma unroll
    for (int mt = 0; mt < 2; ++mt) {
      ah[mt] = load_a_frag_bf16(Ahi, m0 + 16 * mt, k0, lane);
      al[mt] = load_a_frag_bf16(Alo, m0 + 16 * mt, k0, lane);
    }
#pragma unroll
    for (int nt = 0; nt < 2; ++nt) {
      int tc = nt * 16 + (lane & 15);
      bh[nt] = *(const v16bf*)(Bhi + tc * DH + kb);
      bl[nt] = *(const v16bf*)(Blo + tc * DH + kb);
    }
#pragma unroll
    for (int mt = 0; mt < 2; ++mt)
#pragma unroll
      for (int nt = 0; nt < 2; ++nt) acc[mt][nt] = wmma3(acc[mt][nt], ah[mt], al[mt], bh[nt], bl[nt]);
  }

#pragma unroll
  for (int mt = 0; mt < 2; ++mt)
#pragma unroll
    for (int nt = 0; nt < 2; ++nt) {
      int col = n0 + nt * 16 + (lane & 15);
      if (col < VOCAB) {
#pragma unroll
        for (int i = 0; i < 8; ++i) {
          int row = m0 + mt * 16 + i + ((lane >> 4) << 3); // m = step*64 + batch
          int b = row & (BATCH - 1);
          int s = row >> 6;
          Out[((size_t)(b * NDRAFT + s)) * VOCAB + col] = acc[mt][nt][i];
        }
      }
    }
}

// ---------------------------------------------------------------------------
extern "C" void kernel_launch(void* const* d_in, const int* in_sizes, int n_in,
                              void* d_out, int out_size, void* d_ws, size_t ws_size,
                              hipStream_t stream) {
  const float* hidden = (const float*)d_in[0];
  const int*   tids   = (const int*)  d_in[1];
  const float* ip_w   = (const float*)d_in[2];
  const float* ip_b   = (const float*)d_in[3];
  const float* w_ih0  = (const float*)d_in[4];
  const float* w_hh0  = (const float*)d_in[5];
  const float* b_ih0  = (const float*)d_in[6];
  const float* b_hh0  = (const float*)d_in[7];
  const float* w_ih1  = (const float*)d_in[8];
  const float* w_hh1  = (const float*)d_in[9];
  const float* b_ih1  = (const float*)d_in[10];
  const float* b_hh1  = (const float*)d_in[11];
  const float* embed  = (const float*)d_in[12];
  const float* out_w  = (const float*)d_in[13];
  float* out = (float*)d_out;

  float* ws = (float*)d_ws;
  const int ST = BATCH * DH; // 32768
  float*  s0[2] = { ws,          ws + ST };
  float*  s1[2] = { ws + 2 * ST, ws + 3 * ST };
  float*  xAll  = ws + 4 * ST;                 // 4*ST
  __bf16* H2hi  = (__bf16*)(ws + 12 * ST);     // 4*ST bf16 (= 2*ST floats)
  __bf16* H2lo  = (__bf16*)(ws + 14 * ST);     // 4*ST bf16

  k_h0<<<4, 256, 0, stream>>>(hidden, ip_w, ip_b, s0[0], s1[0]);
  k_gather<<<(NDRAFT * BATCH * DH + 255) / 256, 256, 0, stream>>>(embed, tids, xAll);
  for (int s = 0; s < NDRAFT; ++s) {
    int p = s & 1;
    k_gru<<<16, 256, 0, stream>>>(xAll + s * ST, s0[p], w_ih0, w_hh0, b_ih0, b_hh0,
                                  s0[1 - p], nullptr, nullptr);
    k_gru<<<16, 256, 0, stream>>>(s0[1 - p], s1[p], w_ih1, w_hh1, b_ih1, b_hh1,
                                  s1[1 - p], H2hi + s * ST, H2lo + s * ST);
  }
  k_logits<<<(VOCAB + 31) / 32, 256, 32 * DH * 8, stream>>>(H2hi, H2lo, out_w, out);
}
